// PCRNN_2491081031818
// MI455X (gfx1250) — matrix operand order repaired and checked by reading, not compiled, not run on hardware
//
#include <hip/hip_runtime.h>
#include <hip/hip_bf16.h>
#include <cstdint>

// PCRNN forward for MI455X (gfx1250), wave32 + WMMA bf16 + TDM LDS staging.
// S=256, B=16, V=32000, E=512, H=1024.

#define S_LEN 256
#define BATCH 16
#define VOCAB 32000
#define EDIM  512
#define HDIM  1024
#define ROWS  (S_LEN * BATCH)   // 4096
#define BK    64                // K-slab staged per TDM transfer

typedef __attribute__((ext_vector_type(16))) __bf16          v16bf;
typedef __attribute__((ext_vector_type(8)))  float           v8f;
typedef __attribute__((ext_vector_type(8)))  unsigned short  v8us;
typedef __attribute__((ext_vector_type(16))) unsigned short  v16us;
typedef __attribute__((ext_vector_type(4)))  unsigned int    v4u;
typedef __attribute__((ext_vector_type(8)))  int             v8i;
typedef __attribute__((ext_vector_type(4)))  int             v4i;

#if defined(__has_builtin)
#if __has_builtin(__builtin_amdgcn_tensor_load_to_lds) && \
    __has_builtin(__builtin_amdgcn_s_wait_tensorcnt)
#define HAVE_TDM 1
#endif
#endif
#ifndef HAVE_TDM
#define HAVE_TDM 0
#endif

__device__ __forceinline__ unsigned short f2bf(float f) {
  unsigned u = __builtin_bit_cast(unsigned, f);
  unsigned r = u + 0x7FFFu + ((u >> 16) & 1u);   // round-to-nearest-even
  return (unsigned short)(r >> 16);
}

__device__ __forceinline__ v16bf as_v16bf(v8us lo, v8us hi) {
  v16us u = __builtin_shufflevector(lo, hi, 0,1,2,3,4,5,6,7,8,9,10,11,12,13,14,15);
  union { v16us u; v16bf b; } c; c.u = u; return c.b;
}

__device__ __forceinline__ v8f wmma_bf16(v16bf a, v16bf b, v8f c) {
  return __builtin_amdgcn_wmma_f32_16x16x32_bf16(false, a, false, b, (short)0, c,
                                                 false, false);
}

// A fragment: 16x32 bf16, row-major (lda elems). k order used here (contiguous, matches B):
// M=lane&15; lanes<16 hold K 0..15, lanes>=16 hold K 16..31 (contiguous).
__device__ __forceinline__ v16bf load_frag_A(const unsigned short* A, int lda,
                                             int row0, int k0, int lane) {
  const int m  = row0 + (lane & 15);
  const int kk = k0 + ((lane >> 4) << 4);
  const unsigned short* p = A + (size_t)m * lda + kk;
  v8us lo = *reinterpret_cast<const v8us*>(p);
  v8us hi = *reinterpret_cast<const v8us*>(p + 8);
  return as_v16bf(lo, hi);
}

// B fragment: 32x16 where B(k,n) = W[n,k], W row-major NxK (ldw elems).
// n = lane&15; lanes<16 hold K 0..15, lanes>=16 hold K 16..31, pairs K=2v,2v+1.
__device__ __forceinline__ v16bf load_frag_B(const unsigned short* W, int ldw,
                                             int n0, int k0, int lane) {
  const int n  = n0 + (lane & 15);
  const int kk = k0 + ((lane >> 4) << 4);
  const unsigned short* p = W + (size_t)n * ldw + kk;
  v8us lo = *reinterpret_cast<const v8us*>(p);
  v8us hi = *reinterpret_cast<const v8us*>(p + 8);
  return as_v16bf(lo, hi);
}

__device__ __forceinline__ unsigned ldsoff(const void* p) {
  // LDS aperture occupies the high dword of the flat address; low 32 bits are
  // the byte offset the TDM descriptor wants.
  return (unsigned)(uintptr_t)p;
}

#if HAVE_TDM
// One 2-D tile load (tile_d1 rows x tile_d0 elems, 2-byte data, row stride
// stride_elems) global -> LDS via the Tensor Data Mover (D# per §8.3/8.4).
__device__ __forceinline__ void tdm_load_2d(unsigned lds_byte_addr, const void* gaddr,
                                            unsigned tile_d0, unsigned tile_d1,
                                            unsigned stride_elems) {
  unsigned long long ga = (unsigned long long)(uintptr_t)gaddr;
  v4u g0 = { 1u,                                  // count=1, user mode, no gather
             lds_byte_addr,                       // lds_addr
             (unsigned)ga,                        // global_addr[31:0]
             (unsigned)((ga >> 32) & 0x1ffffffu)  // global_addr[56:32]
                 | (2u << 30) };                  // type=2 ("image")
  const unsigned t0 = stride_elems;               // tensor_dim0 (row length)
  const unsigned t1 = 1u << 20;                   // tensor_dim1: large, no OOB clip
  v8i g1;
  g1[0] = (int)(1u << 16);                        // wg_mask=0, data_size=1 (2 bytes)
  g1[1] = (int)((t0 & 0xffffu) << 16);            // barrier_addr=0 | dim0 lo16
  g1[2] = (int)((t0 >> 16) | ((t1 & 0xffffu) << 16));      // dim0 hi | dim1 lo
  g1[3] = (int)((t1 >> 16) | ((tile_d0 & 0xffffu) << 16)); // dim1 hi | tile_dim0
  g1[4] = (int)(tile_d1 & 0xffffu);               // tile_dim1 | tile_dim2=0
  g1[5] = (int)stride_elems;                      // tensor_dim0_stride lo32
  g1[6] = 0;
  g1[7] = 0;
  v4i z4 = {0, 0, 0, 0};
#if __clang_major__ >= 23
  v8i z8 = {0, 0, 0, 0, 0, 0, 0, 0};
  __builtin_amdgcn_tensor_load_to_lds(g0, g1, z4, z4, z8, 0);
#else
  __builtin_amdgcn_tensor_load_to_lds(g0, g1, z4, z4, 0);
#endif
}
#endif

// ---------------- elementwise converts ----------------
__global__ void cvt_f32_bf16_x4(const float* __restrict__ in,
                                unsigned short* __restrict__ out, int n4) {
  int i = blockIdx.x * blockDim.x + threadIdx.x;
  if (i >= n4) return;
  float4 f = reinterpret_cast<const float4*>(in)[i];
  uint2 pk;
  pk.x = (unsigned)f2bf(f.x) | ((unsigned)f2bf(f.y) << 16);
  pk.y = (unsigned)f2bf(f.z) | ((unsigned)f2bf(f.w) << 16);
  reinterpret_cast<uint2*>(out)[i] = pk;
}

__global__ void embed_gather(const int* __restrict__ inputs,
                             const float* __restrict__ emb,
                             unsigned short* __restrict__ Xe) {
  const int r   = blockIdx.x;
  const int tok = inputs[r];
  const float* src = emb + (size_t)tok * EDIM;
  unsigned short* dst = Xe + (size_t)r * EDIM;
  for (int e = threadIdx.x * 4; e < EDIM; e += blockDim.x * 4) {
    float4 f = *reinterpret_cast<const float4*>(src + e);
    uint2 pk;
    pk.x = (unsigned)f2bf(f.x) | ((unsigned)f2bf(f.y) << 16);
    pk.y = (unsigned)f2bf(f.z) | ((unsigned)f2bf(f.w) << 16);
    *reinterpret_cast<uint2*>(dst + e) = pk;
  }
}

// --------- bf16 WMMA GEMM, TDM-staged LDS double buffer -------------------
// C[M,N] = A[M,K] * W[N,K]^T (+bias). Block 256 thr = 8 waves (4M x 2N),
// block tile 128x128, wave tile 32x64, K staged in 128xBK LDS slabs.
__global__ __launch_bounds__(256) void gemm_bf16(const unsigned short* __restrict__ A,
                                                 const unsigned short* __restrict__ W,
                                                 const float* __restrict__ bias,
                                                 float* __restrict__ C,
                                                 int K, int N) {
  __shared__ __align__(16) unsigned short sA[2][128 * BK];   // 16 KB each
  __shared__ __align__(16) unsigned short sB[2][128 * BK];
  const int lane = threadIdx.x & 31;
  const int wave = threadIdx.x >> 5;
  const int wm = wave >> 1;
  const int wn = wave & 1;
  const int m0 = blockIdx.y * 128;
  const int n0 = blockIdx.x * 128;
  const int KT = K / BK;

  v8f acc[2][4] = {};

  // All 12 ds_load_b128 issued up-front per K-sub-step (distinct locals), one
  // dscnt wait, then 8 back-to-back WMMAs (D->C accumulate has no hazard).
  auto compute = [&](int buf) {
#pragma unroll
    for (int kk = 0; kk < BK; kk += 32) {
      v16bf a0 = load_frag_A(sA[buf], BK, wm * 32,      kk, lane);
      v16bf a1 = load_frag_A(sA[buf], BK, wm * 32 + 16, kk, lane);
      v16bf b0 = load_frag_B(sB[buf], BK, wn * 64 +  0, kk, lane);
      v16bf b1 = load_frag_B(sB[buf], BK, wn * 64 + 16, kk, lane);
      v16bf b2 = load_frag_B(sB[buf], BK, wn * 64 + 32, kk, lane);
      v16bf b3 = load_frag_B(sB[buf], BK, wn * 64 + 48, kk, lane);
      acc[0][0] = wmma_bf16(a0, b0, acc[0][0]);
      acc[1][0] = wmma_bf16(a1, b0, acc[1][0]);
      acc[0][1] = wmma_bf16(a0, b1, acc[0][1]);
      acc[1][1] = wmma_bf16(a1, b1, acc[1][1]);
      acc[0][2] = wmma_bf16(a0, b2, acc[0][2]);
      acc[1][2] = wmma_bf16(a1, b2, acc[1][2]);
      acc[0][3] = wmma_bf16(a0, b3, acc[0][3]);
      acc[1][3] = wmma_bf16(a1, b3, acc[1][3]);
    }
  };

#if HAVE_TDM
  if (wave == 0) {
    tdm_load_2d(ldsoff(sA[0]), A + (size_t)m0 * K, BK, 128, K);
    tdm_load_2d(ldsoff(sB[0]), W + (size_t)n0 * K, BK, 128, K);
  }
  for (int kt = 0; kt < KT; ++kt) {
    const int buf = kt & 1;
    if (wave == 0) {
      if (kt + 1 < KT) {
        const int k0 = (kt + 1) * BK;
        tdm_load_2d(ldsoff(sA[buf ^ 1]), A + (size_t)m0 * K + k0, BK, 128, K);
        tdm_load_2d(ldsoff(sB[buf ^ 1]), W + (size_t)n0 * K + k0, BK, 128, K);
        __builtin_amdgcn_s_wait_tensorcnt(2);  // current buffer's pair done
      } else {
        __builtin_amdgcn_s_wait_tensorcnt(0);
      }
    }
    __syncthreads();
    compute(buf);
    __syncthreads();   // buffer free before it is refilled next iteration
  }
#else
  for (int kt = 0; kt < KT; ++kt) {
    const int k0 = kt * BK;
    for (int i = threadIdx.x; i < 128 * BK / 8; i += 256) {
      int r = i >> 3, c = i & 7;   // BK/8 == 8 uint4 per row
      reinterpret_cast<uint4*>(sA[0])[i] =
          reinterpret_cast<const uint4*>(A + (size_t)(m0 + r) * K + k0)[c];
      reinterpret_cast<uint4*>(sB[0])[i] =
          reinterpret_cast<const uint4*>(W + (size_t)(n0 + r) * K + k0)[c];
    }
    __syncthreads();
    compute(0);
    __syncthreads();
  }
#endif

#pragma unroll
  for (int i = 0; i < 2; ++i)
#pragma unroll
    for (int j = 0; j < 4; ++j)
#pragma unroll
      for (int v = 0; v < 8; ++v) {
        int m = m0 + wm * 32 + i * 16 + v + ((lane >> 4) << 3);
        int n = n0 + wn * 64 + j * 16 + (lane & 15);
        float bv = bias ? bias[n] : 0.0f;
        C[(size_t)m * N + n] = acc[i][j][v] + bv;
      }
}

// ---------------- persistent recurrence kernel ----------------
#define RWG   32
#define RCOLS 32

__device__ __forceinline__ void grid_barrier(unsigned* bar, unsigned target) {
  __threadfence();
  __syncthreads();
  if (threadIdx.x == 0) {
    __hip_atomic_fetch_add(bar, 1u, __ATOMIC_ACQ_REL, __HIP_MEMORY_SCOPE_AGENT);
    while (__hip_atomic_load(bar, __ATOMIC_ACQUIRE, __HIP_MEMORY_SCOPE_AGENT) < target) {
      __builtin_amdgcn_s_sleep(8);
    }
  }
  __syncthreads();
}

__global__ __launch_bounds__(64) void rnn_persistent(const float* __restrict__ xbuf,
                                                     const unsigned short* __restrict__ Whh,
                                                     unsigned short* __restrict__ outp,
                                                     unsigned* bar) {
  __shared__ __align__(16) unsigned short sW[RCOLS * HDIM];  // 64 KB
  const int lane = threadIdx.x & 31;
  const int wv   = threadIdx.x >> 5;           // 0..1
  const int n0g  = blockIdx.x * RCOLS + wv * 16;

  // Stage this WG's W_hh slice (rows [blk*32, blk*32+32), contiguous).
#if HAVE_TDM
  if (wv == 0) {
    tdm_load_2d(ldsoff(sW), Whh + (size_t)blockIdx.x * RCOLS * HDIM,
                HDIM, RCOLS, HDIM);
    __builtin_amdgcn_s_wait_tensorcnt(0);
  }
#else
  {
    const uint4* src = reinterpret_cast<const uint4*>(Whh + (size_t)blockIdx.x * RCOLS * HDIM);
    uint4* dst = reinterpret_cast<uint4*>(sW);
    const int nvec = RCOLS * HDIM / 8;
    for (int i = threadIdx.x; i < nvec; i += blockDim.x) dst[i] = src[i];
  }
#endif
  __syncthreads();

  // s == 0: outputs[0] = tanh(x[0]) on our column slice.
  for (int i = threadIdx.x; i < BATCH * RCOLS; i += blockDim.x) {
    int b = i / RCOLS, c = i % RCOLS;
    int col = blockIdx.x * RCOLS + c;
    float v = xbuf[(size_t)b * HDIM + col];
    outp[(size_t)b * HDIM + col] = f2bf(tanhf(v));
  }

  for (int s = 1; s < S_LEN; ++s) {
    grid_barrier(bar, (unsigned)(RWG * s));    // step s-1 fully visible
    const unsigned short* Aptr = outp + (size_t)(s - 1) * BATCH * HDIM;
    v8f acc = {};
    for (int k0 = 0; k0 < HDIM; k0 += 32) {
      v16bf a = load_frag_A(Aptr, HDIM, 0, k0, lane);
      v16bf b = load_frag_B(sW, HDIM, wv * 16, k0, lane);
      acc = wmma_bf16(a, b, acc);
    }
#pragma unroll
    for (int v = 0; v < 8; ++v) {
      int ml  = v + ((lane >> 4) << 3);        // batch row 0..15
      int g   = s * BATCH + ml;
      int col = n0g + (lane & 15);
      float val = acc[v] + xbuf[(size_t)g * HDIM + col];  // h_new
      outp[(size_t)g * HDIM + col] = f2bf(tanhf(val));    // tanh(prediction)
    }
  }
}

extern "C" void kernel_launch(void* const* d_in, const int* in_sizes, int n_in,
                              void* d_out, int out_size, void* d_ws, size_t ws_size,
                              hipStream_t stream) {
  const int*   inputs = (const int*)  d_in[0];   // [S,B]
  const float* emb    = (const float*)d_in[1];   // [V,E]
  const float* W_in   = (const float*)d_in[2];   // [H,E]
  const float* W_hh   = (const float*)d_in[3];   // [H,H]
  const float* W_out  = (const float*)d_in[4];   // [V,H]
  const float* b_out  = (const float*)d_in[5];   // [V]
  float* logits = (float*)d_out;                 // [S*B, V]

  char* ws = (char*)d_ws;
  size_t off = 0;
  auto take = [&](size_t bytes) -> void* {
    void* p = ws + off;
    off = (off + bytes + 255) & ~(size_t)255;
    return p;
  };
  unsigned short* Wout_bf = (unsigned short*)take((size_t)VOCAB * HDIM * 2); // 65.5 MB
  unsigned short* Win_bf  = (unsigned short*)take((size_t)HDIM * EDIM * 2);  // 1 MB
  unsigned short* Whh_bf  = (unsigned short*)take((size_t)HDIM * HDIM * 2);  // 2 MB
  unsigned short* Xe_bf   = (unsigned short*)take((size_t)ROWS * EDIM * 2);  // 4 MB
  float*          xbuf    = (float*)take((size_t)ROWS * HDIM * 4);           // 16 MB
  unsigned short* outp_bf = (unsigned short*)take((size_t)ROWS * HDIM * 2);  // 8 MB
  unsigned*       bar     = (unsigned*)take(256);
  (void)ws_size; (void)in_sizes; (void)n_in; (void)out_size;

  hipMemsetAsync(bar, 0, 256, stream);

  { int n4 = HDIM * EDIM / 4;
    cvt_f32_bf16_x4<<<(n4 + 255) / 256, 256, 0, stream>>>(W_in, Win_bf, n4); }
  { int n4 = HDIM * HDIM / 4;
    cvt_f32_bf16_x4<<<(n4 + 255) / 256, 256, 0, stream>>>(W_hh, Whh_bf, n4); }
  { int n4 = VOCAB * HDIM / 4;
    cvt_f32_bf16_x4<<<(n4 + 255) / 256, 256, 0, stream>>>(W_out, Wout_bf, n4); }

  embed_gather<<<ROWS, 128, 0, stream>>>(inputs, emb, Xe_bf);

  // x = Xe * W_in^T : [4096,1024] f32
  gemm_bf16<<<dim3(HDIM / 128, ROWS / 128), 256, 0, stream>>>(
      Xe_bf, Win_bf, nullptr, xbuf, EDIM, HDIM);

  // Sequential recurrence -> outputs bf16 (= tanh(prediction))
  rnn_persistent<<<RWG, 64, 0, stream>>>(xbuf, Whh_bf, outp_bf, bar);

  // logits = outputs * W_out^T + b_out : [4096, 32000] f32
  gemm_bf16<<<dim3(VOCAB / 128, ROWS / 128), 256, 0, stream>>>(
      outp_bf, Wout_bf, b_out, logits, HDIM, VOCAB);
}
